// SimpleHeadAttentionMultiLatent_75419625717762
// MI455X (gfx1250) — matrix operand order repaired and checked by reading, not compiled, not run on hardware
//
#include <hip/hip_runtime.h>
#include <hip/hip_bf16.h>
#include <stdint.h>

typedef __bf16 bf16;
typedef __attribute__((ext_vector_type(16))) __bf16 v16bf;
typedef __attribute__((ext_vector_type(8)))  __bf16 v8bf;
typedef __attribute__((ext_vector_type(8)))  float  v8f;

#define B_   4
#define S_   2048
#define DM_  256
#define H_   4
#define L_   128
#define HD_  64
#define MROWS (B_*S_)   // 8192

// ---------------- WMMA fragment helpers (16-bit operands, contiguous K in both A and B) ----
// A (16x32, MxK): lanes 0-15 row M=lane; element j -> K = k0+(lane>=16?16:0)+j
//                 (contiguous K, matching B).
__device__ __forceinline__ v16bf load_a_bf16(const bf16* rowbase, int ld, int k0) {
  const int lane = threadIdx.x & 31;
  const int m  = lane & 15;
  const int hi = lane >> 4;
  const bf16* p = rowbase + (size_t)m * ld + k0 + hi * 16;
  v8bf lo = *(const v8bf*)(p);
  v8bf hv = *(const v8bf*)(p + 8);
  v16bf a;
#pragma unroll
  for (int j = 0; j < 8; ++j) { a[j] = lo[j]; a[j + 8] = hv[j]; }
  return a;
}

// B (32x16, KxN) from an N-major (transposed) source: per-lane N = lane&15,
// contiguous K: lanes 0-15 hold K k0..k0+15, lanes 16-31 hold k0+16..k0+31.
__device__ __forceinline__ v16bf load_b_bf16(const bf16* nbase, int ld, int k0) {
  const int lane = threadIdx.x & 31;
  const int n  = lane & 15;
  const int hi = lane >> 4;
  const bf16* p = nbase + (size_t)n * ld + k0 + hi * 16;
  return *(const v16bf*)(p);
}

// A fragment built from fp32 source (LDS or global), converting to bf16.
__device__ __forceinline__ v16bf load_a_f32(const float* rowbase, int ld, int k0) {
  const int lane = threadIdx.x & 31;
  const int m  = lane & 15;
  const int hi = lane >> 4;
  const float* p = rowbase + (size_t)m * ld + k0 + hi * 16;
  v8f lo = *(const v8f*)(p);
  v8f hv = *(const v8f*)(p + 8);
  v16bf a;
#pragma unroll
  for (int j = 0; j < 8; ++j) { a[j] = (bf16)lo[j]; a[j + 8] = (bf16)hv[j]; }
  return a;
}

// ---------------- prep kernels ------------------------------------------------
__global__ void cvt_f32_bf16_kernel(const float* __restrict__ in,
                                    bf16* __restrict__ out, int n) {
  int i = blockIdx.x * blockDim.x + threadIdx.x;
  if (i < n) out[i] = (bf16)in[i];
}

// out[n*K + k] = (bf16) in[k*N + n]   (K x N row-major -> N-major bf16)
__global__ void transpose_cvt_kernel(const float* __restrict__ in,
                                     bf16* __restrict__ out, int K, int N) {
  int i = blockIdx.x * blockDim.x + threadIdx.x;
  if (i >= K * N) return;
  int n = i / K, k = i - n * K;
  out[i] = (bf16)in[(size_t)k * N + n];
}

// ---------------- projections -------------------------------------------------
// q = x@wq_w + wq_b (bf16 out), c_kv = x@w_kv_down (bf16 out). One wave per
// 16x16 tile; blockIdx.y < 32 -> Q tiles, else KV tiles (uniform branch).
__global__ __launch_bounds__(32) void proj_qkv_kernel(
    const bf16* __restrict__ xbf, const bf16* __restrict__ wqT,
    const bf16* __restrict__ kvdT, const float* __restrict__ wq_b,
    bf16* __restrict__ qbf, bf16* __restrict__ ckv)
{
  const int mt = blockIdx.x, nt = blockIdx.y;
  const int lane = threadIdx.x;
  const int m0 = mt * 16;
  const bool isQ = nt < 32;
  const int n0 = isQ ? nt * 16 : (nt - 32) * 16;
  const bf16* wT  = isQ ? wqT : kvdT;
  const bf16* arow = xbf + (size_t)m0 * DM_;
  const bf16* brow = wT  + (size_t)n0 * DM_;

  v8f acc = {};
#pragma unroll
  for (int k0 = 0; k0 < DM_; k0 += 32) {
    v16bf a = load_a_bf16(arow, DM_, k0);
    v16bf b = load_b_bf16(brow, DM_, k0);
    acc = __builtin_amdgcn_wmma_f32_16x16x32_bf16(false, a, false, b,
                                                  (short)0, acc, false, false);
  }
  const int n  = lane & 15;
  const int hi = lane >> 4;
  const float bias = isQ ? wq_b[n0 + n] : 0.0f;
#pragma unroll
  for (int r = 0; r < 8; ++r) {
    const int m = r + hi * 8;
    const float v = acc[r] + bias;
    if (isQ) qbf[(size_t)(m0 + m) * (H_ * L_) + n0 + n] = (bf16)v;
    else     ckv[(size_t)(m0 + m) * L_ + n0 + n]        = (bf16)v;
  }
}

// v = c_kv @ w_v_up, stored TRANSPOSED as vT(B,H,hd,S) for contiguous-K PV loads.
__global__ __launch_bounds__(32) void proj_v_kernel(
    const bf16* __restrict__ ckv, const bf16* __restrict__ vupT,
    bf16* __restrict__ vT)
{
  const int mt = blockIdx.x, nt = blockIdx.y;
  const int lane = threadIdx.x;
  const int m0 = mt * 16, n0 = nt * 16;
  const bf16* arow = ckv  + (size_t)m0 * L_;
  const bf16* brow = vupT + (size_t)n0 * L_;
  v8f acc = {};
#pragma unroll
  for (int k0 = 0; k0 < L_; k0 += 32) {
    v16bf a = load_a_bf16(arow, L_, k0);
    v16bf b = load_b_bf16(brow, L_, k0);
    acc = __builtin_amdgcn_wmma_f32_16x16x32_bf16(false, a, false, b,
                                                  (short)0, acc, false, false);
  }
  const int n  = lane & 15;
  const int hi = lane >> 4;
#pragma unroll
  for (int r = 0; r < 8; ++r) {
    const int g = m0 + r + hi * 8;        // global row in B*S
    const int bb = g >> 11;               // / S_
    const int s  = g & (S_ - 1);
    const int c  = n0 + n;
    const int h  = c >> 6;
    const int d  = c & 63;
    vT[(((size_t)bb * H_ + h) * HD_ + d) * S_ + s] = (bf16)acc[r];
  }
}

// ---------------- fused attention --------------------------------------------
// One workgroup (4 waves) per (b, h, 16-row q tile). The full 16x2048 masked
// logit block lives in LDS (128 KB of the 320 KB WGP LDS).
__global__ __launch_bounds__(128) void attn_kernel(
    const bf16* __restrict__ qbf, const bf16* __restrict__ ckv,
    const bf16* __restrict__ vT, const float* __restrict__ mask,
    float* __restrict__ qk_out, float* __restrict__ aw_out,
    float* __restrict__ attn_ws)
{
  extern __shared__ float smem[];
  float* Llog = smem;                  // 16 * 2048 fp32 masked logits
  float* Red  = smem + 16 * S_;        // 4 * 16 * 64 PV partials
  float* Srow = Red + 4 * 16 * 64;     // 16 * 10 softmax scratch

  const int qt = blockIdx.x, h = blockIdx.y, b = blockIdx.z;
  const int tid  = threadIdx.x;
  const int wave = tid >> 5;
  const int lane = tid & 31;
  const int lm   = lane & 15;
  const int hi   = lane >> 4;
  const float inv_sqrt_dk = 0.08838834764831845f;   // 1/sqrt(128)

  // Q fragments: 16 rows x L=128 -> 4 A-frags (K=32 each)
  const bf16* qrow = qbf + ((size_t)(b * S_) + qt * 16) * (H_ * L_) + h * L_;
  v16bf qa[4];
#pragma unroll
  for (int kc = 0; kc < 4; ++kc) qa[kc] = load_a_bf16(qrow, H_ * L_, kc * 32);

  float mq[8];
#pragma unroll
  for (int r = 0; r < 8; ++r)
    mq[r] = mask[(size_t)b * S_ + qt * 16 + r + hi * 8];

  const size_t qkbase = (((size_t)b * H_ + h) * S_ + (size_t)(qt * 16)) * S_;

  // ---- Phase 1: logits via WMMA; write qk_values; masked logits -> LDS ----
  for (int t = 0; t < 32; ++t) {
    const int kcol0 = (wave * 32 + t) * 16;
    const bf16* krow = ckv + ((size_t)(b * S_) + kcol0 + lm) * L_;
    if (t + 1 < 32)
      __builtin_prefetch(krow + 16 * L_, 0, 1);   // global_prefetch_b8 next tile
    v8f acc = {};
#pragma unroll
    for (int kc = 0; kc < 4; ++kc) {
      v16bf bb = *(const v16bf*)(krow + kc * 32 + hi * 16);
      acc = __builtin_amdgcn_wmma_f32_16x16x32_bf16(false, qa[kc], false, bb,
                                                    (short)0, acc, false, false);
    }
    const int kcol = kcol0 + lm;
    const float mk = mask[(size_t)b * S_ + kcol];
#pragma unroll
    for (int r = 0; r < 8; ++r) {
      const int m = r + hi * 8;
      const float logit = acc[r] * inv_sqrt_dk;
      qk_out[qkbase + (size_t)m * S_ + kcol] = logit;          // pre-mask logits
      const float mm = fminf(1.0f, mq[r] + mk);
      Llog[m * S_ + kcol] = logit + mm * -1e9f;
    }
  }
  __syncthreads();

  // ---- Phase 2: exact softmax over 16 rows in LDS; write attn_weights ----
  {
    const int row = tid >> 3;
    const int sub = tid & 7;
    float* rowp = Llog + row * S_;
    float lmax = -3.4e38f;
    for (int i = 0; i < S_ / 8; ++i) lmax = fmaxf(lmax, rowp[sub + i * 8]);
    Srow[row * 10 + sub] = lmax;
    __syncthreads();
    if (sub == 0) {
      float rm = Srow[row * 10];
      for (int j = 1; j < 8; ++j) rm = fmaxf(rm, Srow[row * 10 + j]);
      Srow[row * 10 + 8] = rm;
    }
    __syncthreads();
    const float rmax = Srow[row * 10 + 8];
    float lsum = 0.0f;
    for (int i = 0; i < S_ / 8; ++i) {
      const int c = sub + i * 8;
      const float e = __expf(rowp[c] - rmax);
      rowp[c] = e;
      lsum += e;
    }
    Srow[row * 10 + sub] = lsum;
    __syncthreads();
    if (sub == 0) {
      float rs = 0.0f;
      for (int j = 0; j < 8; ++j) rs += Srow[row * 10 + j];
      Srow[row * 10 + 9] = rs;
    }
    __syncthreads();
    const float rinv = 1.0f / Srow[row * 10 + 9];
    float* awrow = aw_out + qkbase + (size_t)row * S_;
    for (int i = 0; i < S_ / 8; ++i) {
      const int c = sub + i * 8;
      const float w = rowp[c] * rinv;
      rowp[c] = w;
      awrow[c] = w;
    }
  }
  __syncthreads();

  // ---- Phase 3: PV via WMMA; K=2048 split across 4 waves, reduce in LDS ----
  {
    v8f accs[4] = {};
    const int kstart = wave * 512;
    const bf16* vbase = vT + (((size_t)b * H_ + h) * HD_) * S_;
    for (int it = 0; it < 16; ++it) {
      const int k0 = kstart + it * 32;
      v16bf a = load_a_f32(Llog, S_, k0);               // weights from LDS
#pragma unroll
      for (int ntile = 0; ntile < 4; ++ntile) {
        const bf16* vp = vbase + (size_t)(ntile * 16 + lm) * S_ + k0 + hi * 16;
        v16bf bb = *(const v16bf*)vp;
        accs[ntile] = __builtin_amdgcn_wmma_f32_16x16x32_bf16(
            false, a, false, bb, (short)0, accs[ntile], false, false);
      }
    }
#pragma unroll
    for (int ntile = 0; ntile < 4; ++ntile)
#pragma unroll
      for (int r = 0; r < 8; ++r) {
        const int m = r + hi * 8;
        const int n = ntile * 16 + lm;
        Red[(wave * 16 + m) * 64 + n] = accs[ntile][r];
      }
  }
  __syncthreads();
  for (int e = tid; e < 1024; e += 128) {
    const int m = e >> 6, n = e & 63;
    const float sum = Red[m * 64 + n] + Red[(16 + m) * 64 + n] +
                      Red[(32 + m) * 64 + n] + Red[(48 + m) * 64 + n];
    attn_ws[((size_t)(b * S_) + qt * 16 + m) * DM_ + h * HD_ + n] = sum;
  }
}

// ---------------- output projection ------------------------------------------
__global__ __launch_bounds__(32) void out_proj_kernel(
    const float* __restrict__ attn_ws, const bf16* __restrict__ outT,
    const float* __restrict__ out_b, float* __restrict__ out)
{
  const int mt = blockIdx.x, nt = blockIdx.y;
  const int lane = threadIdx.x;
  const int m0 = mt * 16, n0 = nt * 16;
  const int lm = lane & 15, hi = lane >> 4;
  const float* arow = attn_ws + (size_t)m0 * DM_;
  const bf16*  brow = outT + (size_t)n0 * DM_;
  v8f acc = {};
#pragma unroll
  for (int k0 = 0; k0 < DM_; k0 += 32) {
    v16bf a = load_a_f32(arow, DM_, k0);
    v16bf b = load_b_bf16(brow, DM_, k0);
    acc = __builtin_amdgcn_wmma_f32_16x16x32_bf16(false, a, false, b,
                                                  (short)0, acc, false, false);
  }
  const float bias = out_b[n0 + lm];
#pragma unroll
  for (int r = 0; r < 8; ++r) {
    const int m = r + hi * 8;
    out[(size_t)(m0 + m) * DM_ + n0 + lm] = acc[r] + bias;
  }
}

// ---------------- launcher ----------------------------------------------------
extern "C" void kernel_launch(void* const* d_in, const int* in_sizes, int n_in,
                              void* d_out, int out_size, void* d_ws, size_t ws_size,
                              hipStream_t stream) {
  const float* x       = (const float*)d_in[0];
  const float* mask    = (const float*)d_in[1];
  const float* wq_w    = (const float*)d_in[2];
  const float* wq_b    = (const float*)d_in[3];
  const float* w_kv_dn = (const float*)d_in[4];
  const float* w_v_up  = (const float*)d_in[5];
  const float* out_w   = (const float*)d_in[6];
  const float* out_b   = (const float*)d_in[7];

  float* out = (float*)d_out;
  float* aw  = out + (size_t)MROWS * DM_;                  // attn_weights
  float* qk  = aw  + (size_t)B_ * H_ * S_ * S_;            // qk_values

  uint8_t* w = (uint8_t*)d_ws;
  bf16* xbf  = (bf16*)w;  w += (size_t)MROWS * DM_ * 2;
  bf16* qbf  = (bf16*)w;  w += (size_t)MROWS * H_ * L_ * 2;
  bf16* ckv  = (bf16*)w;  w += (size_t)MROWS * L_ * 2;
  bf16* vT   = (bf16*)w;  w += (size_t)B_ * H_ * HD_ * S_ * 2;
  float* attn = (float*)w; w += (size_t)MROWS * DM_ * 4;
  bf16* wqT  = (bf16*)w;  w += (size_t)DM_ * (H_ * L_) * 2;
  bf16* kvdT = (bf16*)w;  w += (size_t)DM_ * L_ * 2;
  bf16* vupT = (bf16*)w;  w += (size_t)L_ * DM_ * 2;
  bf16* outT = (bf16*)w;  w += (size_t)DM_ * DM_ * 2;

  {
    const int n = MROWS * DM_;
    cvt_f32_bf16_kernel<<<(n + 255) / 256, 256, 0, stream>>>(x, xbf, n);
  }
  transpose_cvt_kernel<<<(DM_*H_*L_ + 255)/256, 256, 0, stream>>>(wq_w,    wqT,  DM_, H_*L_);
  transpose_cvt_kernel<<<(DM_*L_    + 255)/256, 256, 0, stream>>>(w_kv_dn, kvdT, DM_, L_);
  transpose_cvt_kernel<<<(L_*DM_    + 255)/256, 256, 0, stream>>>(w_v_up,  vupT, L_, DM_);
  transpose_cvt_kernel<<<(DM_*DM_   + 255)/256, 256, 0, stream>>>(out_w,   outT, DM_, DM_);

  proj_qkv_kernel<<<dim3(MROWS / 16, 40), 32, 0, stream>>>(xbf, wqT, kvdT, wq_b, qbf, ckv);
  proj_v_kernel  <<<dim3(MROWS / 16, 16), 32, 0, stream>>>(ckv, vupT, vT);

  const size_t shmem = (size_t)(16 * S_ + 4 * 16 * 64 + 16 * 10) * sizeof(float);
  attn_kernel<<<dim3(S_ / 16, H_, B_), 128, shmem, stream>>>(qbf, ckv, vT, mask, qk, aw, attn);

  out_proj_kernel<<<dim3(MROWS / 16, 16), 32, 0, stream>>>(attn, outT, out_b, out);
}
